// DCLM_16329465660010
// MI455X (gfx1250) — matrix operand order repaired and checked by reading, not compiled, not run on hardware
//
#include <hip/hip_runtime.h>
#include <math.h>
#include <stdint.h>

// ---------------- problem constants ----------------
#define BB     4
#define SS     2048
#define HH     1152
#define NHH    12
#define HDD    96
#define NMM    4
#define MDD    128
#define KW     4          // conv kernel taps
#define INNER  4608
#define BT     (BB*SS)    // 8192 tokens

// ---------------- WMMA / TDM types ----------------
typedef __attribute__((ext_vector_type(16))) __bf16 bf16x16;
typedef __attribute__((ext_vector_type(8)))  float  floatx8;
typedef __attribute__((ext_vector_type(4)))  unsigned int u32x4;
typedef __attribute__((ext_vector_type(8)))  int          i32x8;
typedef __attribute__((ext_vector_type(4)))  int          i32x4;

union FragU { uint4 q[2]; bf16x16 v; };

#if __has_builtin(__builtin_amdgcn_tensor_load_to_lds)
#define USE_TDM 1
#else
#define USE_TDM 0
#endif

__device__ __forceinline__ unsigned short f2bf(float f) {
  union { float f; unsigned u; } v; v.f = f;
  unsigned r = v.u + 0x7FFFu + ((v.u >> 16) & 1u);   // round-to-nearest-even
  return (unsigned short)(r >> 16);
}
__device__ __forceinline__ float sigm(float x) { return 1.f / (1.f + expf(-x)); }
__device__ __forceinline__ float gelu_exact(float x) {
  return 0.5f * x * (1.f + erff(x * 0.70710678118654752f));
}

#if USE_TDM
// Issue one 2D bf16 tile load (rows x rowlen elems, row stride in elems) into LDS via TDM.
// Descriptor per CDNA5 ISA ch.8 (D# groups 0/1; groups 2/3 zero => dims >1 unused).
__device__ __forceinline__ void tdm_load_2d(unsigned lds_off, const unsigned short* gptr,
                                            int rows, int rowlen, long long stride_elems,
                                            int tensor_d0, int tensor_d1)
{
  unsigned long long ga = (unsigned long long)(uintptr_t)gptr;
  u32x4 g0 = {0u, 0u, 0u, 0u};
  g0[0] = 1u;                                            // count=1, user descriptor
  g0[1] = lds_off;                                       // lds_addr (bytes)
  g0[2] = (unsigned)(ga & 0xFFFFFFFFull);                // global_addr[31:0]
  g0[3] = (unsigned)((ga >> 32) & 0x1FFFFFFull) | (2u << 30);  // global_addr[56:32] | type=2
  unsigned ull = (unsigned long long)stride_elems & 0xFFFFFFFFull;
  unsigned uhh = (unsigned)(((unsigned long long)stride_elems >> 32) & 0xFFFFull);
  i32x8 g1 = {0, 0, 0, 0, 0, 0, 0, 0};
  g1[0] = (int)(1u << 16);                               // data_size=1 -> 2 bytes/elem
  g1[1] = (int)(((unsigned)tensor_d0 & 0xFFFFu) << 16);  // tensor_dim0[15:0]
  g1[2] = (int)(((unsigned)tensor_d0 >> 16) | (((unsigned)tensor_d1 & 0xFFFFu) << 16));
  g1[3] = (int)(((unsigned)tensor_d1 >> 16) | ((unsigned)rowlen << 16));   // tile_dim0
  g1[4] = (int)((unsigned)rows);                         // tile_dim1 (tile_dim2=0)
  g1[5] = (int)ull;                                      // tensor_dim0_stride[31:0]
  g1[6] = (int)uhh;                                      // stride[47:32] (dim1_stride=0)
  i32x4 gz = {0, 0, 0, 0};
#if __clang_major__ >= 23
  i32x8 gz8 = {0, 0, 0, 0, 0, 0, 0, 0};
  __builtin_amdgcn_tensor_load_to_lds(g0, g1, gz, gz, gz8, 0);
#else
  __builtin_amdgcn_tensor_load_to_lds(g0, g1, gz, gz, 0);
#endif
}
__device__ __forceinline__ void wait_tensor0() {
#if __has_builtin(__builtin_amdgcn_s_wait_tensorcnt)
  __builtin_amdgcn_s_wait_tensorcnt(0);
#endif
}
#endif // USE_TDM

// ---------------- precision pre-passes ----------------
__global__ void cvt_bf16_kernel(const float* __restrict__ in, unsigned short* __restrict__ out,
                                long n)
{
  long idx = (long)blockIdx.x * blockDim.x + threadIdx.x;
  if (idx < n) out[idx] = f2bf(in[idx]);
}

// Wt[n*K + k] = bf16(W[k*N + n])  (coalesced read side)
__global__ void transpose_bf16_kernel(const float* __restrict__ W, unsigned short* __restrict__ Wt,
                                      int Kd, int Nd)
{
  long idx = (long)blockIdx.x * blockDim.x + threadIdx.x;
  if (idx >= (long)Kd * Nd) return;
  int n = (int)(idx % Nd), k = (int)(idx / Nd);
  Wt[(size_t)n * Kd + k] = f2bf(W[idx]);
}

// ---------------- tiled bf16 WMMA GEMM with TDM double-buffered staging ----------------
// C[M,N] = A[M,K] * W[K,N] (+bias); A pre-converted bf16 [M,K]; Wt bf16 transposed [N,K].
constexpr int TBM = 128, TBN = 128, TBK = 32;

__global__ __launch_bounds__(256)
void gemm_wmma_kernel(const unsigned short* __restrict__ Abf,
                      const unsigned short* __restrict__ Wt,
                      const float* __restrict__ bias, float* __restrict__ C,
                      int M, int N, int K)
{
  __shared__ __align__(16) unsigned short Asl[2][TBM * TBK];  // [m][k]
  __shared__ __align__(16) unsigned short Bsl[2][TBN * TBK];  // [n][k]

  const int tid  = threadIdx.x;
  const int n0   = blockIdx.x * TBN;
  const int m0   = blockIdx.y * TBM;
  const int wave = tid >> 5;
  const int lane = tid & 31;
  const int wm   = wave >> 1;     // 0..3 -> 32-row slab
  const int wn   = wave & 1;      // 0..1 -> 64-col slab
  const int hl   = lane >> 4;
  const int l16  = lane & 15;

  floatx8 acc[2][4] = {};
  const int nsteps = K / TBK;

#if USE_TDM
  if (wave == 0) {
    tdm_load_2d((unsigned)(uintptr_t)&Asl[0][0], Abf + (size_t)m0 * K, TBM, TBK, K, K, M);
    tdm_load_2d((unsigned)(uintptr_t)&Bsl[0][0], Wt  + (size_t)n0 * K, TBN, TBK, K, K, N);
    wait_tensor0();
  }
  __syncthreads();
#endif

  for (int step = 0; step < nsteps; ++step) {
    const int cur = step & 1;
    const int k0  = step * TBK;

#if USE_TDM
    // prefetch next tiles into the other buffer while we compute (async DMA)
    if (wave == 0 && step + 1 < nsteps) {
      tdm_load_2d((unsigned)(uintptr_t)&Asl[cur ^ 1][0],
                  Abf + (size_t)m0 * K + (k0 + TBK), TBM, TBK, K, K, M);
      tdm_load_2d((unsigned)(uintptr_t)&Bsl[cur ^ 1][0],
                  Wt  + (size_t)n0 * K + (k0 + TBK), TBN, TBK, K, K, N);
    }
#else
    // cooperative staging fallback: 256 threads copy both 8KB tiles
    {
      const int row = tid >> 1, seg = tid & 1;
      const size_t abase = (size_t)(m0 + row) * K + k0 + seg * 16;
      const size_t bbase = (size_t)(n0 + row) * K + k0 + seg * 16;
      *(uint4*)&Asl[cur][row * TBK + seg * 16]     = *(const uint4*)(Abf + abase);
      *(uint4*)&Asl[cur][row * TBK + seg * 16 + 8] = *(const uint4*)(Abf + abase + 8);
      *(uint4*)&Bsl[cur][row * TBK + seg * 16]     = *(const uint4*)(Wt + bbase);
      *(uint4*)&Bsl[cur][row * TBK + seg * 16 + 8] = *(const uint4*)(Wt + bbase + 8);
    }
    __syncthreads();
#endif

    // ---- fragments + WMMA (ISA 7.12.2 layouts) ----
    FragU fa[2], fb[4];
    #pragma unroll
    for (int mf = 0; mf < 2; ++mf) {
      int r = wm * 32 + mf * 16 + l16;
      // lanes 0-15: K 0..7 & 16..23 ; lanes 16-31: K 8..15 & 24..31
      fa[mf].q[0] = *(const uint4*)&Asl[cur][r * TBK + 8 * hl];
      fa[mf].q[1] = *(const uint4*)&Asl[cur][r * TBK + 16 + 8 * hl];
    }
    #pragma unroll
    for (int nf = 0; nf < 4; ++nf) {
      int cn = wn * 64 + nf * 16 + l16;
      // lanes 0-15: K 0..7 & 16..23 ; lanes 16-31: K 8..15 & 24..31
      fb[nf].q[0] = *(const uint4*)&Bsl[cur][cn * TBK + 8 * hl];
      fb[nf].q[1] = *(const uint4*)&Bsl[cur][cn * TBK + 16 + 8 * hl];
    }
    #pragma unroll
    for (int mf = 0; mf < 2; ++mf)
      #pragma unroll
      for (int nf = 0; nf < 4; ++nf)
        acc[mf][nf] = __builtin_amdgcn_wmma_f32_16x16x32_bf16(
            false, fa[mf].v, false, fb[nf].v, (short)0, acc[mf][nf], false, false);

#if USE_TDM
    if (wave == 0) wait_tensor0();   // next buffer landed
    __syncthreads();
#else
    __syncthreads();
#endif
  }

  // ---- epilogue: C/D layout col = lane%16, row = vgpr + 8*(lane/16) ----
  #pragma unroll
  for (int mf = 0; mf < 2; ++mf)
    #pragma unroll
    for (int nf = 0; nf < 4; ++nf) {
      int col = n0 + wn * 64 + nf * 16 + l16;
      float bv = bias ? bias[col] : 0.f;
      #pragma unroll
      for (int v = 0; v < 8; ++v) {
        int row = m0 + wm * 32 + mf * 16 + v + 8 * hl;
        C[(size_t)row * N + col] = acc[mf][nf][v] + bv;
      }
    }
}

// ---------------- rmsnorm ----------------
__global__ __launch_bounds__(256)
void rmsnorm_kernel(const float* __restrict__ x, const float* __restrict__ w,
                    float* __restrict__ out)
{
  int t = blockIdx.x;
  const float* xr = x + (size_t)t * HH;
  __shared__ float red[256];
  float s = 0.f;
  for (int c = threadIdx.x; c < HH; c += 256) { float v = xr[c]; s += v * v; }
  red[threadIdx.x] = s; __syncthreads();
  for (int off = 128; off > 0; off >>= 1) {
    if ((int)threadIdx.x < off) red[threadIdx.x] += red[threadIdx.x + off];
    __syncthreads();
  }
  float n = rsqrtf(red[0] * (1.f / HH) + 1e-6f);
  for (int c = threadIdx.x; c < HH; c += 256)
    out[(size_t)t * HH + c] = xr[c] * n * w[c];
}

// ---------------- token gate: g[t] = sigmoid(x[t,:].w + b) ----------------
__global__ void tokengate_kernel(const float* __restrict__ x, const float* __restrict__ w,
                                 const float* __restrict__ b, float* __restrict__ g)
{
  int t = blockIdx.x * blockDim.x + threadIdx.x;
  if (t >= BT) return;
  const float* xr = x + (size_t)t * HH;
  float s = b[0];
  for (int c = 0; c < HH; ++c) s += xr[c] * w[c];
  g[t] = sigm(s);
}

// ---------------- conv-stack layer: out = h + gelu(dconv(h)) ----------------
__global__ void convstack_kernel(const float* __restrict__ h, const float* __restrict__ w,
                                 const float* __restrict__ b, float* __restrict__ out, int d)
{
  int idx = blockIdx.x * blockDim.x + threadIdx.x;
  if (idx >= BT * HH) return;
  int c  = idx % HH;
  int s  = (idx / HH) % SS;
  int bb = idx / (HH * SS);
  float acc = b[c];
  #pragma unroll
  for (int j = 0; j < KW; ++j) {
    int tpos = s - (KW - 1 - j) * d;
    if (tpos >= 0) acc += w[c * KW + j] * h[((size_t)bb * SS + tpos) * HH + c];
  }
  out[idx] = h[idx] + gelu_exact(acc);
}

// ---------------- per-head conv layer (no activation) ----------------
__constant__ int c_head_dil[NHH * 3] = {
  1,2,4,  1,1,1,  4,8,16,  8,16,32,  32,64,128,  64,128,256,
  256,512,1024,  1,100,200,  1,500,1000,  1,1024,2048,  3,9,27,  5,25,125 };

__global__ void headconv_kernel(const float* __restrict__ h, const float* __restrict__ hw,
                                const float* __restrict__ hb, float* __restrict__ out, int j)
{
  int idx = blockIdx.x * blockDim.x + threadIdx.x;
  if (idx >= BT * HH) return;
  int d  = idx % HDD;
  int i  = (idx / HDD) % NHH;
  int s  = (idx / HH) % SS;
  int bb = idx / (HH * SS);
  int dil = c_head_dil[i * 3 + j];
  float acc = hb[((size_t)i * 3 + j) * HDD + d];
  #pragma unroll
  for (int jj = 0; jj < KW; ++jj) {
    int tpos = s - (KW - 1 - jj) * dil;
    if (tpos >= 0)
      acc += hw[(((size_t)i * 3 + j) * HDD + d) * KW + jj] *
             h[(((size_t)bb * SS + tpos) * NHH + i) * HDD + d];
  }
  out[idx] = h[idx] + acc;
}

// ---------------- router: hw[t,i] = sigmoid(x[t,:].rw[:,i] + rb[i]) ----------------
__global__ void router_kernel(const float* __restrict__ x, const float* __restrict__ rw,
                              const float* __restrict__ rb, float* __restrict__ hw)
{
  int idx = blockIdx.x * blockDim.x + threadIdx.x;
  if (idx >= BT * NHH) return;
  int i = idx % NHH, t = idx / NHH;
  float s = rb[i];
  const float* xr = x + (size_t)t * HH;
  for (int c = 0; c < HH; ++c) s += xr[c] * rw[c * NHH + i];
  hw[idx] = sigm(s);
}

// ---------------- GLU gate: out[t,c] = in[t,c] * sigmoid(in[t,c+width]) ----------------
__global__ void glugate_kernel(const float* __restrict__ in, float* __restrict__ out,
                               int width, int pitch)
{
  int idx = blockIdx.x * blockDim.x + threadIdx.x;
  if (idx >= BT * width) return;
  int c = idx % width, t = idx / width;
  float a = in[(size_t)t * pitch + c];
  float b = in[(size_t)t * pitch + width + c];
  out[idx] = a * sigm(b);
}

// ---------------- neural-memory projections ----------------
__global__ void memproj_kernel(const float* __restrict__ xh, const float* __restrict__ W,
                               float* __restrict__ out, int E)
{
  int idx = blockIdx.x * blockDim.x + threadIdx.x;
  if (idx >= BT * NMM * E) return;
  int e = idx % E, n = (idx / E) % NMM, t = idx / (E * NMM);
  const float* xm = xh + ((size_t)t * NHH + 6 + n) * HDD;
  const float* wr = W + (size_t)n * HDD * E;
  float s = 0.f;
  for (int d = 0; d < HDD; ++d) s += xm[d] * wr[d * E + e];
  out[idx] = s;
}

__global__ void memgate_kernel(const float* __restrict__ xh, const float* __restrict__ gw,
                               const float* __restrict__ gb, float* __restrict__ g)
{
  int idx = blockIdx.x * blockDim.x + threadIdx.x;
  if (idx >= BT * NMM) return;
  int n = idx % NMM, t = idx / NMM;
  const float* xm = xh + ((size_t)t * NHH + 6 + n) * HDD;
  float s = gb[n];
  for (int d = 0; d < HDD; ++d) s += xm[d] * gw[n * HDD + d];
  g[idx] = sigm(s);
}

// ---------------- chunked fast-weight scan; one block per (b,n) ----------------
#define STRIDE_C 64
__global__ __launch_bounds__(256)
void memscan_kernel(const float* __restrict__ qb, const float* __restrict__ kb,
                    const float* __restrict__ vb, const float* __restrict__ gb,
                    float* __restrict__ reads)
{
  int bb = blockIdx.x / NMM, n = blockIdx.x % NMM;
  __shared__ float Ms[HDD * MDD];   // 48 KB fast-weight state
  for (int i = threadIdx.x; i < HDD * MDD; i += 256) Ms[i] = 0.f;
  __syncthreads();

  for (int c = 0; c < SS / STRIDE_C; ++c) {
    int sbase = c * STRIDE_C;
    float ga = 0.f;
    for (int s = 0; s < STRIDE_C; ++s)
      ga += gb[(size_t)(bb * SS + sbase + s) * NMM + n];
    ga *= (1.f / STRIDE_C);

    // reads with PRE-update state
    for (int o = threadIdx.x; o < STRIDE_C * MDD; o += 256) {
      int s = o / MDD, m = o % MDD;
      const float* qr = qb + ((size_t)(bb * SS + sbase + s) * NMM + n) * HDD;
      float acc = 0.f;
      for (int d = 0; d < HDD; ++d) acc += qr[d] * Ms[d * MDD + m];
      reads[((size_t)(bb * SS + sbase + s) * NMM + n) * MDD + m] = acc;
    }
    __syncthreads();

    // state update: M = (1-ga)*M + (k*g)^T v
    for (int o = threadIdx.x; o < HDD * MDD; o += 256) {
      int d = o / MDD, m = o % MDD;
      float wsum = 0.f;
      for (int s = 0; s < STRIDE_C; ++s) {
        size_t ti = (size_t)(bb * SS + sbase + s);
        float kk = kb[(ti * NMM + n) * HDD + d] * gb[ti * NMM + n];
        wsum += kk * vb[(ti * NMM + n) * MDD + m];
      }
      Ms[o] = (1.f - ga) * Ms[o] + wsum;
    }
    __syncthreads();
  }
}

// ---------------- memory readout ----------------
__global__ void memout_kernel(const float* __restrict__ reads, const float* __restrict__ Wout,
                              float* __restrict__ memo)
{
  int idx = blockIdx.x * blockDim.x + threadIdx.x;
  if (idx >= BT * NMM * HDD) return;
  int d = idx % HDD, n = (idx / HDD) % NMM, t = idx / (HDD * NMM);
  const float* rr = reads + ((size_t)t * NMM + n) * MDD;
  const float* wr = Wout + (size_t)n * MDD * HDD;
  float s = 0.f;
  for (int m = 0; m < MDD; ++m) s += rr[m] * wr[m * HDD + d];
  memo[idx] = s;
}

// ---------------- combine heads: (hstate + mem) * head_weight ----------------
__global__ void combine_kernel(const float* __restrict__ hstate, const float* __restrict__ memo,
                               const float* __restrict__ hw, float* __restrict__ out)
{
  int idx = blockIdx.x * blockDim.x + threadIdx.x;
  if (idx >= BT * HH) return;
  int c = idx % HH, t = idx / HH;
  int i = c / HDD, d = c % HDD;
  float v = hstate[idx];
  if (i >= 6 && i <= 9) v += memo[((size_t)t * NMM + (i - 6)) * HDD + d];
  out[idx] = v * hw[(size_t)t * NHH + i];
}

// ---------------- elementwise helpers ----------------
__global__ void residualgate_kernel(const float* __restrict__ xin, const float* __restrict__ g,
                                    const float* __restrict__ v, float* __restrict__ out, int n)
{
  int idx = blockIdx.x * blockDim.x + threadIdx.x;
  if (idx >= n) return;
  int t = idx / HH;
  out[idx] = xin[idx] + g[t] * v[idx];
}

__global__ void mulsig_kernel(const float* __restrict__ u, const float* __restrict__ v,
                              float* __restrict__ out, int n)
{
  int idx = blockIdx.x * blockDim.x + threadIdx.x;
  if (idx >= n) return;
  out[idx] = u[idx] * sigm(v[idx]);
}

// ---------------- host orchestration ----------------
extern "C" void kernel_launch(void* const* d_in, const int* in_sizes, int n_in,
                              void* d_out, int out_size, void* d_ws, size_t ws_size,
                              hipStream_t stream)
{
  const float* x           = (const float*)d_in[0];
  const float* norm1_w     = (const float*)d_in[1];
  const float* norm2_w     = (const float*)d_in[2];
  const float* norm3_w     = (const float*)d_in[3];
  const float* cs_w        = (const float*)d_in[4];
  const float* cs_b        = (const float*)d_in[5];
  const float* cs_proj_w   = (const float*)d_in[6];
  const float* cs_proj_b   = (const float*)d_in[7];
  const float* gate_proj_w = (const float*)d_in[8];
  const float* router_w    = (const float*)d_in[9];
  const float* router_b    = (const float*)d_in[10];
  const float* head_w      = (const float*)d_in[11];
  const float* head_b      = (const float*)d_in[12];
  const float* mem_q       = (const float*)d_in[13];
  const float* mem_k       = (const float*)d_in[14];
  const float* mem_v       = (const float*)d_in[15];
  const float* mem_gw      = (const float*)d_in[16];
  const float* mem_gb      = (const float*)d_in[17];
  const float* mem_out     = (const float*)d_in[18];
  const float* mixgate_w   = (const float*)d_in[19];
  const float* mixgate_b   = (const float*)d_in[20];
  const float* mixing_w    = (const float*)d_in[21];
  const float* mixing_b    = (const float*)d_in[22];
  const float* ffn_in_w    = (const float*)d_in[23];
  const float* ffn_out_w   = (const float*)d_in[24];
  const float* cg_w        = (const float*)d_in[25];
  const float* cg_b        = (const float*)d_in[26];
  const float* sg_w        = (const float*)d_in[27];
  const float* sg_b        = (const float*)d_in[28];
  const float* fg_w        = (const float*)d_in[29];
  const float* fg_b        = (const float*)d_in[30];

  // bump allocator over d_ws (bytes)
  size_t off = 0;
  auto allocB = [&](size_t bytes) -> void* {
    void* p = (char*)d_ws + off;
    off += ((bytes + 255) / 256) * 256;
    return p;
  };
  float* buf0  = (float*)allocB((size_t)BT * 9216 * 4);   // widest GEMM outputs
  float* buf1  = (float*)allocB((size_t)BT * INNER * 4);  // hsw / mixgate logits / out2
  float* buf2  = (float*)allocB((size_t)BT * HH * 4);     // running residual x
  float* buf3  = (float*)allocB((size_t)BT * HH * 4);     // norms / conv ping
  float* buf4  = (float*)allocB((size_t)BT * HH * 4);     // conv pong / gemm outs
  float* g0    = (float*)allocB((size_t)BT * 4);
  float* hwbuf = (float*)allocB((size_t)BT * NHH * 4);
  float* qb    = (float*)allocB((size_t)BT * NMM * HDD * 4);
  float* kb    = (float*)allocB((size_t)BT * NMM * HDD * 4);
  float* vb    = (float*)allocB((size_t)BT * NMM * MDD * 4);
  float* gmem  = (float*)allocB((size_t)BT * NMM * 4);
  float* rdbuf = (float*)allocB((size_t)BT * NMM * MDD * 4);
  float* memo  = (float*)allocB((size_t)BT * NMM * HDD * 4);
  // bf16 staging: activations + transposed weights
  unsigned short* abf      = (unsigned short*)allocB((size_t)BT * INNER * 2);
  unsigned short* wt_cspj  = (unsigned short*)allocB((size_t)HH * HH * 2);
  unsigned short* wt_gate  = (unsigned short*)allocB((size_t)2 * HH * HH * 2);
  unsigned short* wt_mixg  = (unsigned short*)allocB((size_t)HH * HH * 2);
  unsigned short* wt_mix   = (unsigned short*)allocB((size_t)HH * HH * 2);
  unsigned short* wt_fin   = (unsigned short*)allocB((size_t)HH * 2 * INNER * 2);
  unsigned short* wt_fout  = (unsigned short*)allocB((size_t)INNER * HH * 2);
  (void)ws_size; (void)in_sizes; (void)n_in; (void)out_size;

  const int TH = BT * HH;
  const int EB = 256;
  auto blocks = [](long n) { return (int)((n + 255) / 256); };

  // one-time (per call) weight transpose+convert to bf16 [N][K]
  auto transp = [&](const float* W, unsigned short* Wt, int Kd, int Nd) {
    long n = (long)Kd * Nd;
    transpose_bf16_kernel<<<blocks(n), EB, 0, stream>>>(W, Wt, Kd, Nd);
  };
  transp(cs_proj_w,   wt_cspj, HH, HH);
  transp(gate_proj_w, wt_gate, HH, 2 * HH);
  transp(mixgate_w,   wt_mixg, HH, HH);
  transp(mixing_w,    wt_mix,  HH, HH);
  transp(ffn_in_w,    wt_fin,  HH, 2 * INNER);
  transp(ffn_out_w,   wt_fout, INNER, HH);

  auto gemm = [&](const float* A, const unsigned short* Wt, const float* bias, float* C,
                  int M, int N, int K) {
    long na = (long)M * K;
    cvt_bf16_kernel<<<blocks(na), EB, 0, stream>>>(A, abf, na);
    dim3 grid(N / TBN, M / TBM);
    gemm_wmma_kernel<<<grid, dim3(256), 0, stream>>>(abf, Wt, bias, C, M, N, K);
  };

  // ===== Phase 1: conv stack =====
  tokengate_kernel<<<blocks(BT), EB, 0, stream>>>(x, cg_w, cg_b, g0);
  rmsnorm_kernel<<<BT, 256, 0, stream>>>(x, norm1_w, buf3);
  {
    const int dil[6] = {1, 2, 4, 8, 16, 32};
    float* pa = buf3; float* pb = buf4;
    for (int i = 0; i < 6; ++i) {
      convstack_kernel<<<blocks(TH), EB, 0, stream>>>(
          pa, cs_w + (size_t)i * HH * KW, cs_b + (size_t)i * HH, pb, dil[i]);
      float* t = pa; pa = pb; pb = t;
    }   // result in buf3
  }
  gemm(buf3, wt_cspj, cs_proj_b, buf4, BT, HH, HH);
  residualgate_kernel<<<blocks(TH), EB, 0, stream>>>(x, g0, buf4, buf2, TH);   // x1

  // ===== Phase 2: multi-head state =====
  tokengate_kernel<<<blocks(BT), EB, 0, stream>>>(buf2, sg_w, sg_b, g0);
  rmsnorm_kernel<<<BT, 256, 0, stream>>>(buf2, norm2_w, buf3);
  router_kernel<<<blocks((long)BT * NHH), EB, 0, stream>>>(buf3, router_w, router_b, hwbuf);
  gemm(buf3, wt_gate, nullptr, buf0, BT, 2 * HH, HH);
  glugate_kernel<<<blocks(TH), EB, 0, stream>>>(buf0, buf4, HH, 2 * HH);       // xg/xh

  memproj_kernel<<<blocks((long)BT * NMM * HDD), EB, 0, stream>>>(buf4, mem_q, qb, HDD);
  memproj_kernel<<<blocks((long)BT * NMM * HDD), EB, 0, stream>>>(buf4, mem_k, kb, HDD);
  memproj_kernel<<<blocks((long)BT * NMM * MDD), EB, 0, stream>>>(buf4, mem_v, vb, MDD);
  memgate_kernel<<<blocks((long)BT * NMM), EB, 0, stream>>>(buf4, mem_gw, mem_gb, gmem);
  memscan_kernel<<<BB * NMM, 256, 0, stream>>>(qb, kb, vb, gmem, rdbuf);
  memout_kernel<<<blocks((long)BT * NMM * HDD), EB, 0, stream>>>(rdbuf, mem_out, memo);

  headconv_kernel<<<blocks(TH), EB, 0, stream>>>(buf4, head_w, head_b, buf3, 0);
  headconv_kernel<<<blocks(TH), EB, 0, stream>>>(buf3, head_w, head_b, buf4, 1);
  headconv_kernel<<<blocks(TH), EB, 0, stream>>>(buf4, head_w, head_b, buf3, 2); // hstate

  combine_kernel<<<blocks(TH), EB, 0, stream>>>(buf3, memo, hwbuf, buf4);
  gemm(buf4, wt_mixg, mixgate_b, buf1, BT, HH, HH);
  mulsig_kernel<<<blocks(TH), EB, 0, stream>>>(buf4, buf1, buf1, TH);
  gemm(buf1, wt_mix, mixing_b, buf4, BT, HH, HH);
  residualgate_kernel<<<blocks(TH), EB, 0, stream>>>(buf2, g0, buf4, buf2, TH); // x2

  // ===== Phase 3: FFN (GLU) =====
  tokengate_kernel<<<blocks(BT), EB, 0, stream>>>(buf2, fg_w, fg_b, g0);
  rmsnorm_kernel<<<BT, 256, 0, stream>>>(buf2, norm3_w, buf3);
  gemm(buf3, wt_fin, nullptr, buf0, BT, 2 * INNER, HH);
  glugate_kernel<<<blocks((long)BT * INNER), EB, 0, stream>>>(buf0, buf1, INNER, 2 * INNER);
  gemm(buf1, wt_fout, nullptr, buf4, BT, HH, INNER);
  residualgate_kernel<<<blocks(TH), EB, 0, stream>>>(buf2, g0, buf4, (float*)d_out, TH);
}
